// MLPAttnEdgeAggregation_25529285607946
// MI455X (gfx1250) — matrix operand order repaired and checked by reading, compiled, not run on hardware
//
#include <hip/hip_runtime.h>
#include <hip/hip_bf16.h>

typedef __attribute__((ext_vector_type(16))) _Float16 v16h;
typedef __attribute__((ext_vector_type(8)))  _Float16 v8h;
typedef __attribute__((ext_vector_type(8)))  float    v8f;

#define N_TOK 8192
#define NPG   512
#define NB    16
#define MNB   32
#define DM    256
#define DE    128
#define NH    8
#define VLD   320   // vcat leading dim: 256 v | 8 qa | 8 ka | pad to 320

// ---------------------------------------------------------------- reductions
__device__ __forceinline__ float red8(float v){
  v += __shfl_xor(v, 1, 32);
  v += __shfl_xor(v, 2, 32);
  v += __shfl_xor(v, 4, 32);
  return v;
}
__device__ __forceinline__ float red32(float v){
  #pragma unroll
  for (int off = 1; off < 32; off <<= 1) v += __shfl_xor(v, off, 32);
  return v;
}

// ---------------------------------------------------------------- weight prep
__global__ __launch_bounds__(256) void prep_weights_kernel(
    const float* __restrict__ Wqkv, const float* __restrict__ bqkv,
    const float* __restrict__ wattn,
    const float* __restrict__ Wqe,  const float* __restrict__ bqe,
    const float* __restrict__ wedge,
    const float* __restrict__ Wfc1, const float* __restrict__ Wfc2,
    _Float16* __restrict__ Wcat, float* __restrict__ bcat,
    float* __restrict__ weffE, float* __restrict__ biasEA,
    _Float16* __restrict__ Wfc1h, _Float16* __restrict__ Wfc2h)
{
  const int T0 = 256*VLD, T1 = VLD, T2 = DE*NH, T3 = NH, T4 = 384*256, T5 = 256*256;
  const int total = T0+T1+T2+T3+T4+T5;
  for (int idx = blockIdx.x*blockDim.x + threadIdx.x; idx < total;
       idx += gridDim.x*blockDim.x){
    int i = idx;
    if (i < T0){
      int c = i / VLD, j = i % VLD; float v;
      if (j < 256)      v = Wqkv[c*768 + 512 + j];                      // W_v
      else if (j < 264){ int h = j-256; v = 0.f;                        // wq_eff
        for (int d=0; d<32; d++) v += Wqkv[c*768 + h*32 + d]*wattn[d]; }
      else if (j < 272){ int h = j-264; v = 0.f;                        // wk_eff
        for (int d=0; d<32; d++) v += Wqkv[c*768 + 256 + h*32 + d]*wattn[d]; }
      else v = 0.f;
      Wcat[i] = (_Float16)v; continue;
    }
    i -= T0;
    if (i < T1){
      float v;
      if (i < 256)      v = bqkv[512 + i];
      else if (i < 264){ int h = i-256; v = 0.f;
        for (int d=0; d<32; d++) v += bqkv[h*32 + d]*wattn[d]; }
      else if (i < 272){ int h = i-264; v = 0.f;
        for (int d=0; d<32; d++) v += bqkv[256 + h*32 + d]*wattn[d]; }
      else v = 0.f;
      bcat[i] = v; continue;
    }
    i -= T1;
    if (i < T2){                 // weffE[c*8+h] = sum_e Wqe[c, h*16+e]*wedge[e]
      int c = i >> 3, h = i & 7; float v = 0.f;
      for (int e=0; e<16; e++) v += Wqe[c*256 + h*16 + e]*wedge[e];
      weffE[i] = v; continue;
    }
    i -= T2;
    if (i < T3){
      float v = 0.f;
      for (int e=0; e<16; e++) v += bqe[i*16 + e]*wedge[e];
      biasEA[i] = v; continue;
    }
    i -= T3;
    if (i < T4){ Wfc1h[i] = (_Float16)Wfc1[i]; continue; }
    i -= T4;
    Wfc2h[i] = (_Float16)Wfc2[i];
  }
}

// ---------------------------------------------------------------- geometry
__global__ __launch_bounds__(256) void geom_kernel(
    const float* __restrict__ geo, float* __restrict__ center,
    float* __restrict__ Vout, float* __restrict__ proj)
{
  __shared__ float sred[256];
  __shared__ float sStat[9];
  __shared__ float sC[3];
  __shared__ float sV[9];
  const int b = blockIdx.x, tid = threadIdx.x;
  const float* X = geo + (size_t)b*NPG*3;

  float q[3] = {0.f,0.f,0.f};
  for (int p = tid; p < NPG; p += 256){
    q[0] += X[p*3+0]; q[1] += X[p*3+1]; q[2] += X[p*3+2];
  }
  for (int i=0;i<3;i++){
    sred[tid] = q[i]; __syncthreads();
    for (int st=128; st>0; st>>=1){ if (tid<st) sred[tid]+=sred[tid+st]; __syncthreads(); }
    if (tid==0) sStat[i] = sred[0];
    __syncthreads();
  }
  if (tid<3) sC[tid] = sStat[tid] * (1.f/NPG);
  __syncthreads();
  const float cx = sC[0], cy = sC[1], cz = sC[2];

  float c6[6] = {0.f,0.f,0.f,0.f,0.f,0.f};
  for (int p = tid; p < NPG; p += 256){
    float dx = X[p*3]-cx, dy = X[p*3+1]-cy, dz = X[p*3+2]-cz;
    c6[0]+=dx*dx; c6[1]+=dx*dy; c6[2]+=dx*dz; c6[3]+=dy*dy; c6[4]+=dy*dz; c6[5]+=dz*dz;
  }
  for (int i=0;i<6;i++){
    sred[tid] = c6[i]; __syncthreads();
    for (int st=128; st>0; st>>=1){ if (tid<st) sred[tid]+=sred[tid+st]; __syncthreads(); }
    if (tid==0) sStat[i] = sred[0];
    __syncthreads();
  }
  if (tid==0){
    float Am[3][3] = {{sStat[0],sStat[1],sStat[2]},
                      {sStat[1],sStat[3],sStat[4]},
                      {sStat[2],sStat[4],sStat[5]}};
    float Vm[3][3] = {{1,0,0},{0,1,0},{0,0,1}};
    for (int sweep=0; sweep<12; sweep++){
      for (int pi=0; pi<3; pi++){
        const int p  = (pi==2) ? 1 : 0;
        const int qq = (pi==0) ? 1 : 2;
        float apq = Am[p][qq];
        if (fabsf(apq) > 1e-20f){
          float th = 0.5f*atan2f(2.f*apq, Am[qq][qq]-Am[p][p]);
          float c = cosf(th), s = sinf(th);
          for (int k=0;k<3;k++){ float a0=Am[k][p], a1=Am[k][qq];
            Am[k][p] = c*a0 - s*a1; Am[k][qq] = s*a0 + c*a1; }
          for (int k=0;k<3;k++){ float a0=Am[p][k], a1=Am[qq][k];
            Am[p][k] = c*a0 - s*a1; Am[qq][k] = s*a0 + c*a1; }
          for (int k=0;k<3;k++){ float v0=Vm[k][p], v1=Vm[k][qq];
            Vm[k][p] = c*v0 - s*v1; Vm[k][qq] = s*v0 + c*v1; }
        }
      }
    }
    for (int i=0;i<3;i++) for (int j=0;j<3;j++){
      sV[i*3+j] = Vm[i][j]; Vout[b*9 + i*3 + j] = Vm[i][j];
    }
    center[b*3+0]=cx; center[b*3+1]=cy; center[b*3+2]=cz;
  }
  __syncthreads();
  for (int p = tid; p < NPG; p += 256){
    float dx = X[p*3]-cx, dy = X[p*3+1]-cy, dz = X[p*3+2]-cz;
    size_t gp = (size_t)b*NPG + p;
    #pragma unroll
    for (int j=0;j<3;j++)
      proj[gp*3+j] = dx*sV[0+j] + dy*sV[3+j] + dz*sV[6+j];
  }
}

// ---------------------------------------------------------------- token LN + gate
__global__ __launch_bounds__(256) void token_pre_kernel(
    const float* __restrict__ tok,
    const float* __restrict__ lng, const float* __restrict__ lnb,
    const float* __restrict__ Wg,  const float* __restrict__ bg,
    _Float16* __restrict__ lnout, float* __restrict__ gate)
{
  const int wave = threadIdx.x >> 5, lane = threadIdx.x & 31;
  const int row = blockIdx.x*8 + wave;
  const float* x = tok + (size_t)row*DM;
  float xs[8]; float s = 0.f, sq = 0.f, gd = 0.f;
  #pragma unroll
  for (int i=0;i<8;i++){ int c = lane + 32*i; float v = x[c];
    xs[i]=v; s+=v; sq+=v*v; gd += v*Wg[c]; }
  s = red32(s); sq = red32(sq); gd = red32(gd);
  const float mean = s*(1.f/DM);
  const float var  = sq*(1.f/DM) - mean*mean;
  const float rs   = rsqrtf(var + 1e-5f);
  #pragma unroll
  for (int i=0;i<8;i++){ int c = lane + 32*i;
    lnout[(size_t)row*DM + c] = (_Float16)((xs[i]-mean)*rs*lng[c] + lnb[c]); }
  if (lane==0) gate[row] = 1.f/(1.f + expf(-(gd + bg[0])));
}

// ---------------------------------------------------------------- row LN f32->f16
__global__ __launch_bounds__(256) void rowln_kernel(
    const float* __restrict__ Xin,
    const float* __restrict__ lng, const float* __restrict__ lnb,
    _Float16* __restrict__ outh)
{
  const int wave = threadIdx.x >> 5, lane = threadIdx.x & 31;
  const int row = blockIdx.x*8 + wave;
  const float* x = Xin + (size_t)row*DM;
  float xs[8]; float s = 0.f, sq = 0.f;
  #pragma unroll
  for (int i=0;i<8;i++){ int c = lane + 32*i; float v = x[c];
    xs[i]=v; s+=v; sq+=v*v; }
  s = red32(s); sq = red32(sq);
  const float mean = s*(1.f/DM);
  const float var  = sq*(1.f/DM) - mean*mean;
  const float rs   = rsqrtf(var + 1e-5f);
  #pragma unroll
  for (int i=0;i<8;i++){ int c = lane + 32*i;
    outh[(size_t)row*DM + c] = (_Float16)((xs[i]-mean)*rs*lng[c] + lnb[c]); }
}

// ---------------------------------------------------------------- WMMA GEMM
// C[M,N] = A[M,K](f16) x B[K,N](f16) + bias (+gelu) (+residual), f32 accum.
// Compile-time K/strides -> fully unrolled, software-pipelined fragments.
// Block = 8 waves, wave tile 16x64 (1 A-frag x 4 B-frags), block tile 128x64.
template<int K, int LDA, int LDB, int LDO, bool GELU, bool RES>
__global__ __launch_bounds__(256) void wmma_gemm_kernel(
    const _Float16* __restrict__ A, const _Float16* __restrict__ B,
    const float* __restrict__ bias, const float* __restrict__ residual,
    float* __restrict__ outF, _Float16* __restrict__ outH)
{
  const int wave = threadIdx.x >> 5;
  const int lane = threadIdx.x & 31;
  const int r  = lane & 15;
  const int hi = lane >> 4;
  const int m0 = blockIdx.x * 128;
  const int n0 = blockIdx.y * 64;

  // A fragment: lane(r,hi) holds row m0+wave*16+r, K = {kk+8hi+0..7, kk+16+8hi+0..7}
  const _Float16* pa = A + (size_t)(m0 + wave*16 + r)*LDA + 8*hi;
  // B fragments: lane(r,hi) holds column n0+t*16+r, K = {kk+8hi+0..7, kk+16+8hi+0..7}
  const _Float16* pb = B + (size_t)(8*hi)*LDB + n0 + r;

  auto loadA = [&](int kk)->v16h{
    v8h lo = *(const v8h*)(pa + kk);
    v8h hp = *(const v8h*)(pa + kk + 16);
    return __builtin_shufflevector(lo, hp, 0,1,2,3,4,5,6,7,8,9,10,11,12,13,14,15);
  };
  auto loadB = [&](int kk, int t)->v16h{
    v16h v;
    #pragma unroll
    for (int j=0;j<8;j++){
      v[j]   = pb[(size_t)(kk+j)*LDB + t*16];
      v[8+j] = pb[(size_t)(kk+16+j)*LDB + t*16];
    }
    return v;
  };

  constexpr int NK = K/32;
  v8f acc[4] = {};
  v16h ac = loadA(0);
  v16h bc0 = loadB(0,0), bc1 = loadB(0,1), bc2 = loadB(0,2), bc3 = loadB(0,3);
  #pragma unroll
  for (int i=0; i<NK; i++){
    v16h an{}, bn0{}, bn1{}, bn2{}, bn3{};
    if (i+1 < NK){                      // prefetch next chunk before compute
      an  = loadA((i+1)*32);
      bn0 = loadB((i+1)*32,0); bn1 = loadB((i+1)*32,1);
      bn2 = loadB((i+1)*32,2); bn3 = loadB((i+1)*32,3);
    }
    acc[0] = __builtin_amdgcn_wmma_f32_16x16x32_f16(false, ac, false, bc0, (short)0, acc[0], false, false);
    acc[1] = __builtin_amdgcn_wmma_f32_16x16x32_f16(false, ac, false, bc1, (short)0, acc[1], false, false);
    acc[2] = __builtin_amdgcn_wmma_f32_16x16x32_f16(false, ac, false, bc2, (short)0, acc[2], false, false);
    acc[3] = __builtin_amdgcn_wmma_f32_16x16x32_f16(false, ac, false, bc3, (short)0, acc[3], false, false);
    ac = an; bc0 = bn0; bc1 = bn1; bc2 = bn2; bc3 = bn3;
  }
  #pragma unroll
  for (int t=0;t<4;t++){
    #pragma unroll
    for (int g=0; g<8; g++){
      const int row = m0 + wave*16 + g + 8*hi;
      const int col = n0 + t*16 + r;
      float v = acc[t][g];
      v += bias[col];
      if (GELU) v = 0.5f*v*(1.f + erff(v*0.70710678118654752f));  // exact GELU
      if (RES)  v += residual[(size_t)row*LDO + col];
      if (outF) outF[(size_t)row*LDO + col] = v;
      else      outH[(size_t)row*LDO + col] = (_Float16)v;
    }
  }
}

// ---------------------------------------------------------------- fused edge LN + attention + aggregation
// One workgroup (256 thr) per token. The token's contiguous 16 KB edge tile is
// staged global->LDS with async b128 copies (ASYNCcnt), then LN'd from LDS.
__global__ __launch_bounds__(256) void edge_attn_kernel(
    const float* __restrict__ edge,
    const int* __restrict__ nidx,
    const unsigned char* __restrict__ nmask,
    const float* __restrict__ lne_g, const float* __restrict__ lne_b,
    const float* __restrict__ weffE, const float* __restrict__ biasEA,
    const float* __restrict__ Wqe,   const float* __restrict__ bqe,
    const float* __restrict__ Wfa,   const float* __restrict__ bfa,
    const float* __restrict__ vcat,
    const float* __restrict__ proj,
    const float* __restrict__ center, const float* __restrict__ Vmat,
    const float* __restrict__ gate,   const float* __restrict__ geoin,
    _Float16* __restrict__ hcat, float* __restrict__ geoout)
{
  __shared__ float4 sRaw4[MNB*DE/4];           // raw edge tile (async dest), 16 KB
  __shared__ __align__(16) float sLN[MNB*DE];  // LN'd edge feats, 16 KB
  __shared__ float sWe[DE*NH];
  __shared__ float sLg[DE], sLb[DE];
  __shared__ float sEA[MNB*NH];
  __shared__ float sAttn[NH*MNB];
  __shared__ float sAh[NH*DE];
  __shared__ int   sNbr[MNB];
  __shared__ int   sMask[MNB];
  __shared__ float sGeo[NH*3];
  __shared__ float sT[3];

  const int n = blockIdx.x;
  const int tid = threadIdx.x;
  const int m = tid >> 3, s = tid & 7;   // neighbor / sub-thread (8 per neighbor)

  // ---- async-stage this token's 32x128 f32 edge tile into LDS (4x b128/lane)
  {
    const uint64_t gbase = (uint64_t)(uintptr_t)(edge + (size_t)n*MNB*DE);
    const uint32_t lbase = (uint32_t)(uintptr_t)(void*)sRaw4;  // LDS byte offset
    #pragma unroll
    for (int it=0; it<4; it++){
      const uint32_t boff = (uint32_t)(it*256 + tid)*16u;
      uint32_t ldst = lbase + boff;
      uint64_t gsrc = gbase + boff;
      asm volatile("global_load_async_to_lds_b128 %0, %1, off"
                   :: "v"(ldst), "v"(gsrc) : "memory");
    }
  }

  for (int i = tid; i < DE*NH; i += 256) sWe[i] = weffE[i];
  if (tid < DE){ sLg[tid] = lne_g[tid]; sLb[tid] = lne_b[tid]; }
  if (s == 0){
    int nb = nidx[n*MNB + m];
    sNbr[m] = nb; sMask[m] = nmask[n*MNB + m];
    __builtin_prefetch(&vcat[(size_t)nb*VLD], 0, 0);   // warm gathered v rows
  }
  asm volatile("s_wait_asynccnt 0x0" ::: "memory");    // own wave's copies done
  __syncthreads();                                     // all waves' copies done

  // ---- LayerNorm: one neighbor per 8 threads, 16 contiguous channels each
  const float* sRaw = (const float*)sRaw4;
  float xs[16]; float ps = 0.f, pq = 0.f;
  #pragma unroll
  for (int i=0;i<4;i++){
    float4 v = sRaw4[(m*DE + s*16)/4 + i];
    xs[4*i+0]=v.x; xs[4*i+1]=v.y; xs[4*i+2]=v.z; xs[4*i+3]=v.w;
    ps += v.x+v.y+v.z+v.w;
    pq += v.x*v.x + v.y*v.y + v.z*v.z + v.w*v.w;
  }
  ps = red8(ps); pq = red8(pq);
  const float mean = ps*(1.f/DE);
  const float var  = pq*(1.f/DE) - mean*mean;
  const float rs   = rsqrtf(var + 1e-5f);
  float pea[NH] = {0,0,0,0,0,0,0,0};
  #pragma unroll
  for (int i=0;i<4;i++){
    float4 y4;
    #pragma unroll
    for (int j=0;j<4;j++){
      const int c = s*16 + 4*i + j;
      const float y = (xs[4*i+j]-mean)*rs*sLg[c] + sLb[c];
      (&y4.x)[j] = y;
      #pragma unroll
      for (int h=0;h<NH;h++) pea[h] += y * sWe[c*NH + h];
    }
    *(float4*)(&sLN[m*DE + s*16 + 4*i]) = y4;
  }
  #pragma unroll
  for (int h=0;h<NH;h++) pea[h] = red8(pea[h]);
  if (s == 0){
    #pragma unroll
    for (int h=0;h<NH;h++) sEA[m*NH + h] = pea[h];
  }
  __syncthreads();

  // ---- softmax over 32 neighbors, one thread per head
  if (tid < NH){
    const int h = tid;
    const float qa = vcat[(size_t)n*VLD + 256 + h] + biasEA[h];
    float lv[MNB]; float mx = -3.4e38f;
    #pragma unroll
    for (int mm=0; mm<MNB; mm++){
      float l = qa + vcat[(size_t)sNbr[mm]*VLD + 264 + h] + sEA[mm*NH + h];
      if (!sMask[mm]) l = -1e9f;
      lv[mm] = l; mx = fmaxf(mx, l);
    }
    float sum = 0.f;
    #pragma unroll
    for (int mm=0; mm<MNB; mm++){ float e = expf(lv[mm]-mx); lv[mm]=e; sum+=e; }
    const float inv = 1.f/sum;
    #pragma unroll
    for (int mm=0; mm<MNB; mm++) sAttn[h*MNB + mm] = lv[mm]*inv;
  }
  __syncthreads();

  // ---- attn-weighted aggregations
  for (int idx = tid; idx < NH*DE; idx += 256){   // A_h[h][c]
    const int h = idx >> 7, c = idx & 127;
    float acc = 0.f;
    #pragma unroll
    for (int mm=0; mm<MNB; mm++) acc += sAttn[h*MNB+mm]*sLN[mm*DE + c];
    sAh[idx] = acc;
  }
  { // scalar_ctx: hcat[:, tid] with tid = h*32+d
    const int h = tid >> 5;
    float acc = 0.f;
    #pragma unroll
    for (int mm=0; mm<MNB; mm++) acc += sAttn[h*MNB+mm]*vcat[(size_t)sNbr[mm]*VLD + tid];
    hcat[(size_t)n*384 + tid] = (_Float16)acc;
  }
  if (tid < NH*3){   // geo per-head aggregate
    const int h = tid/3, j = tid - 3*h;
    float acc = 0.f;
    #pragma unroll
    for (int mm=0; mm<MNB; mm++) acc += sAttn[h*MNB+mm]*proj[(size_t)sNbr[mm]*3 + j];
    sGeo[tid] = acc;
  }
  __syncthreads();

  if (tid < DE){     // edge_ctx = A_h @ W_ve + b_ve
    const int h = tid >> 4, e = tid & 15;
    float acc = bqe[DE + h*16 + e];
    for (int c=0; c<DE; c++) acc += sAh[h*DE + c]*Wqe[(size_t)c*256 + DE + h*16 + e];
    hcat[(size_t)n*384 + 256 + tid] = (_Float16)acc;
  }
  if (tid < 3){      // collapsed frame average: odd part of silu
    const int j = tid; float u = 0.f;
    #pragma unroll
    for (int h=0;h<NH;h++) u += sGeo[h*3 + j]*Wfa[h];
    const float bf = bfa[0];
    const float up = u + bf, un = -u + bf;
    sT[j] = 0.5f*( up/(1.f+expf(-up)) - un/(1.f+expf(-un)) );
  }
  __syncthreads();
  if (tid < 3){
    const int i = tid, b = n >> 9;
    float gc = center[b*3 + i];
    #pragma unroll
    for (int j=0;j<3;j++) gc += Vmat[b*9 + i*3 + j]*sT[j];
    const float gt = gate[n];
    geoout[(size_t)n*3 + i] = gc*gt + geoin[(size_t)n*3 + i]*(1.f - gt);
  }
}

// ---------------------------------------------------------------- host launcher
extern "C" void kernel_launch(void* const* d_in, const int* in_sizes, int n_in,
                              void* d_out, int out_size, void* d_ws, size_t ws_size,
                              hipStream_t stream)
{
  const float* tok   = (const float*)d_in[0];
  const float* geo   = (const float*)d_in[1];
  const float* edge  = (const float*)d_in[2];
  const int*   nidx  = (const int*)d_in[3];
  /* d_in[4] batch_idx unused (derivable as n/512) */
  const unsigned char* nmask = (const unsigned char*)d_in[5];
  const float* ln_qg = (const float*)d_in[6];
  const float* ln_qb = (const float*)d_in[7];
  const float* Wqkv  = (const float*)d_in[8];
  const float* bqkv  = (const float*)d_in[9];
  const float* ln_eg = (const float*)d_in[10];
  const float* ln_eb = (const float*)d_in[11];
  const float* Wqe   = (const float*)d_in[12];
  const float* bqe   = (const float*)d_in[13];
  const float* wattn = (const float*)d_in[14];
  const float* wedge = (const float*)d_in[15];
  const float* Wg    = (const float*)d_in[16];
  const float* bg    = (const float*)d_in[17];
  const float* Wfc1  = (const float*)d_in[18];
  const float* bfc1  = (const float*)d_in[19];
  const float* ln_hg = (const float*)d_in[20];
  const float* ln_hb = (const float*)d_in[21];
  const float* Wfc2  = (const float*)d_in[22];
  const float* bfc2  = (const float*)d_in[23];
  const float* Wfa   = (const float*)d_in[24];
  const float* bfa   = (const float*)d_in[25];
  (void)in_sizes; (void)n_in; (void)out_size; (void)ws_size;

  char* ws = (char*)d_ws;
  size_t off = 0;
  auto alloc = [&](size_t bytes)->size_t{
    size_t r = off; off += (bytes + 255) & ~(size_t)255; return r; };

  size_t oWcat  = alloc((size_t)256*VLD*sizeof(_Float16));
  size_t oBcat  = alloc((size_t)VLD*sizeof(float));
  size_t oWeffE = alloc((size_t)DE*NH*sizeof(float));
  size_t oBea   = alloc((size_t)NH*sizeof(float));
  size_t oWfc1h = alloc((size_t)384*256*sizeof(_Float16));
  size_t oWfc2h = alloc((size_t)256*256*sizeof(_Float16));
  size_t oCen   = alloc((size_t)NB*3*sizeof(float));
  size_t oV     = alloc((size_t)NB*9*sizeof(float));
  size_t oProj  = alloc((size_t)N_TOK*3*sizeof(float));
  size_t oLNt   = alloc((size_t)N_TOK*DM*sizeof(_Float16));
  size_t oGate  = alloc((size_t)N_TOK*sizeof(float));
  size_t oVcat  = alloc((size_t)N_TOK*VLD*sizeof(float));
  size_t oHcat  = alloc((size_t)N_TOK*384*sizeof(_Float16));
  size_t oH1    = alloc((size_t)N_TOK*DM*sizeof(float));
  size_t oHmid  = alloc((size_t)N_TOK*DM*sizeof(_Float16));

  _Float16* Wcat  = (_Float16*)(ws + oWcat);
  float*    Bcat  = (float*)(ws + oBcat);
  float*    WeffE = (float*)(ws + oWeffE);
  float*    Bea   = (float*)(ws + oBea);
  _Float16* Wfc1h = (_Float16*)(ws + oWfc1h);
  _Float16* Wfc2h = (_Float16*)(ws + oWfc2h);
  float*    Cen   = (float*)(ws + oCen);
  float*    Vm    = (float*)(ws + oV);
  float*    Proj  = (float*)(ws + oProj);
  _Float16* LNt   = (_Float16*)(ws + oLNt);
  float*    Gate  = (float*)(ws + oGate);
  float*    Vcat  = (float*)(ws + oVcat);
  _Float16* Hcat  = (_Float16*)(ws + oHcat);
  float*    H1    = (float*)(ws + oH1);
  _Float16* Hmid  = (_Float16*)(ws + oHmid);

  float* scalar_out = (float*)d_out;
  float* geo_out    = scalar_out + (size_t)N_TOK*DM;

  prep_weights_kernel<<<256, 256, 0, stream>>>(
      Wqkv, bqkv, wattn, Wqe, bqe, wedge, Wfc1, Wfc2,
      Wcat, Bcat, WeffE, Bea, Wfc1h, Wfc2h);

  geom_kernel<<<NB, 256, 0, stream>>>(geo, Cen, Vm, Proj);

  token_pre_kernel<<<N_TOK/8, 256, 0, stream>>>(
      tok, ln_qg, ln_qb, Wg, bg, LNt, Gate);

  // vcat = LN(tok) @ Wcat + bcat : [v | qa | ka | pad]   (8192 x 320, K=256)
  wmma_gemm_kernel<256, 256, VLD, VLD, false, false>
      <<<dim3(N_TOK/128, VLD/64), 256, 0, stream>>>(
      LNt, Wcat, Bcat, nullptr, Vcat, nullptr);

  edge_attn_kernel<<<N_TOK, 256, 0, stream>>>(
      edge, nidx, nmask, ln_eg, ln_eb, WeffE, Bea, Wqe, bqe, Wfa, bfa,
      Vcat, Proj, Cen, Vm, Gate, geo, Hcat, geo_out);

  // h1 = gelu(hcat @ W_fc1 + b_fc1)   (8192 x 256, K=384)
  wmma_gemm_kernel<384, 384, 256, 256, true, false>
      <<<dim3(N_TOK/128, 256/64), 256, 0, stream>>>(
      Hcat, Wfc1h, bfc1, nullptr, H1, nullptr);

  rowln_kernel<<<N_TOK/8, 256, 0, stream>>>(H1, ln_hg, ln_hb, Hmid);

  // scalar_out = hmid @ W_fc2 + b_fc2 + token_embs   (8192 x 256, K=256)
  wmma_gemm_kernel<256, 256, 256, 256, false, true>
      <<<dim3(N_TOK/128, 256/64), 256, 0, stream>>>(
      Hmid, Wfc2h, bfc2, tok, scalar_out, nullptr);
}
